// DecoderBlock_54288386621890
// MI455X (gfx1250) — matrix operand order repaired and checked by reading, not compiled, not run on hardware
//
#include <hip/hip_runtime.h>

// ---------------------------------------------------------------------------
// Types for CDNA5 (gfx1250) WMMA: wave32, v_wmma_f32_16x16x32_bf16
// ---------------------------------------------------------------------------
typedef __bf16 bf16;
typedef __attribute__((ext_vector_type(16))) bf16  v16bf;
typedef __attribute__((ext_vector_type(8)))  float v8f;
typedef __attribute__((ext_vector_type(4)))  unsigned int u32x4;
typedef int v4i __attribute__((vector_size(16)));   // matches builtin's pointee type

union FragAB { v16bf v; u32x4 q[2]; bf16 h[16]; };

#define WMMA_BF16(a, b, c) \
  __builtin_amdgcn_wmma_f32_16x16x32_bf16(false, (a), false, (b), (short)0, (c), false, false)

// ---------------------------------------------------------------------------
// Async global->LDS staging (GLOBAL_LOAD_ASYNC_TO_LDS_B128, ASYNCcnt) with a
// synchronous fallback if the builtin is not available in this toolchain.
// ---------------------------------------------------------------------------
#if defined(__has_builtin)
#if __has_builtin(__builtin_amdgcn_global_load_async_to_lds_b128)
#define HAS_ASYNC_LDS 1
#endif
#endif
#ifndef HAS_ASYNC_LDS
#define HAS_ASYNC_LDS 0
#endif

__device__ __forceinline__ void stage_b128(const bf16* g, bf16* l) {
#if HAS_ASYNC_LDS
  typedef __attribute__((address_space(1))) v4i* gp_t;
  typedef __attribute__((address_space(3))) v4i* lp_t;
  __builtin_amdgcn_global_load_async_to_lds_b128(
      (gp_t)(__attribute__((address_space(1))) void*)(void*)g,
      (lp_t)(__attribute__((address_space(3))) void*)(void*)l,
      0, 0);
#else
  *(u32x4*)l = *(const u32x4*)g;
#endif
}

__device__ __forceinline__ void stage_wait() {
#if HAS_ASYNC_LDS
  asm volatile("s_wait_asynccnt 0" ::: "memory");
#endif
}

// ---------------------------------------------------------------------------
// f32 -> bf16 conversion
// ---------------------------------------------------------------------------
__global__ void cvt_f32_bf16(const float* __restrict__ in, bf16* __restrict__ out, int n) {
  int i = blockIdx.x * 256 + threadIdx.x;
  if (i < n) out[i] = (bf16)in[i];
}

// ---------------------------------------------------------------------------
// GEMM: C[M,N] = A[M,K](bf16) @ W[K,N](bf16) + bias, optional ReLU,
// output either f32 (Cf) or bf16 (Cb).
// Block tile 128x128, BK=64, 8 waves (4M x 2N), each wave 32x64 = 2x4 tiles,
// 16 WMMAs per K-step. Double-buffered async-to-LDS pipeline: tile i+1 is in
// flight (ASYNCcnt) while tile i is consumed by WMMAs.
// ---------------------------------------------------------------------------
template <bool RELU, bool OUTBF>
__global__ __launch_bounds__(256) void gemm_bias_kernel(
    const bf16* __restrict__ A, const bf16* __restrict__ Bw,
    const float* __restrict__ bias, float* __restrict__ Cf, bf16* __restrict__ Cb,
    int M, int N, int K) {
  constexpr int LDA = 72;               // A row stride (144B, 16B aligned)
  constexpr int LDB = 136;              // B row stride (272B, 16B aligned)
  __shared__ bf16 As[2][128 * LDA];     // [m][k] row-major, double buffered
  __shared__ bf16 Bs[2][64 * LDB];      // [k][n] row-major, double buffered

  const int tid  = threadIdx.x;
  const int lane = tid & 31, wave = tid >> 5;
  const int lh = lane >> 4, ln = lane & 15;
  const int wm = wave >> 1, wn = wave & 1;
  const int m0 = blockIdx.y * 128, n0 = blockIdx.x * 128;

  const int arow = tid >> 1, ac0 = (tid & 1) * 32;   // A loader mapping
  const int bkr  = tid >> 2, bc0 = (tid & 3) * 32;   // B loader mapping

  auto issue_tiles = [&](int kt, int buf) {
    const bf16* asrc = A + (size_t)(m0 + arow) * K + kt + ac0;
    bf16* adst = &As[buf][arow * LDA + ac0];
#pragma unroll
    for (int j = 0; j < 4; j++) stage_b128(asrc + j * 8, adst + j * 8);
    const bf16* bsrc = Bw + (size_t)(kt + bkr) * N + n0 + bc0;
    bf16* bdst = &Bs[buf][bkr * LDB + bc0];
#pragma unroll
    for (int j = 0; j < 4; j++) stage_b128(bsrc + j * 8, bdst + j * 8);
  };

  v8f zero = {};
  v8f acc[2][4];
#pragma unroll
  for (int i = 0; i < 2; i++)
#pragma unroll
    for (int j = 0; j < 4; j++) acc[i][j] = zero;

  issue_tiles(0, 0);
  stage_wait();
  __syncthreads();

  int buf = 0;
  for (int kt = 0; kt < K; kt += 64) {
    if (kt + 64 < K) issue_tiles(kt + 64, buf ^ 1);  // overlap with compute

    // A fragments: [mt][dc], contiguous 16B LDS reads (A lane row = ln)
    FragAB af[2][2];
#pragma unroll
    for (int mt = 0; mt < 2; mt++)
#pragma unroll
      for (int dc = 0; dc < 2; dc++) {
        int r = wm * 32 + mt * 16 + ln;
        const bf16* ap = &As[buf][r * LDA + dc * 32];
        af[mt][dc].q[0] = *(const u32x4*)(ap + 16 * lh);       // k = 16*lh..+7
        af[mt][dc].q[1] = *(const u32x4*)(ap + 16 * lh + 8);  // k = 16*lh+8..+15
      }

    // B fragments gathered column-wise from row-major tile; chain WMMAs
#pragma unroll
    for (int nt = 0; nt < 4; nt++) {
      int col = wn * 64 + nt * 16 + ln;   // B lane col = ln (both halves)
#pragma unroll
      for (int dc = 0; dc < 2; dc++) {
        FragAB bfr;
        const bf16* bp = &Bs[buf][(dc * 32 + 16 * lh) * LDB + col];  // k = 16*lh + e
#pragma unroll
        for (int e = 0; e < 16; e++) bfr.h[e] = bp[e * LDB];
#pragma unroll
        for (int mt = 0; mt < 2; mt++)
          acc[mt][nt] = WMMA_BF16(af[mt][dc].v, bfr.v, acc[mt][nt]);
      }
    }
    stage_wait();
    __syncthreads();
    buf ^= 1;
  }

  // Epilogue: C layout row = r + 8*laneHi, col = laneLo
#pragma unroll
  for (int mt = 0; mt < 2; mt++) {
#pragma unroll
    for (int nt = 0; nt < 4; nt++) {
      int col = n0 + wn * 64 + nt * 16 + ln;
      float bc = bias[col];
#pragma unroll
      for (int r = 0; r < 8; r++) {
        int row = m0 + wm * 32 + mt * 16 + r + 8 * lh;
        float v = acc[mt][nt][r] + bc;
        if (RELU) v = fmaxf(v, 0.0f);
        if (OUTBF) Cb[(size_t)row * N + col] = (bf16)v;
        else       Cf[(size_t)row * N + col] = v;
      }
    }
  }
}

// ---------------------------------------------------------------------------
// Flash-style attention, faithful to reference:
//   scores = floor(Q K^T / 8); (mask is a no-op in the reference); softmax; P V
// Grid (S/64, H, B), 128 threads = 4 waves, each wave owns a 16-query tile.
// K/V tiles staged row-major via double-buffered async-to-LDS pipeline.
// ---------------------------------------------------------------------------
__global__ __launch_bounds__(128) void attn_kernel(
    const bf16* __restrict__ Q, const bf16* __restrict__ K,
    const bf16* __restrict__ V, bf16* __restrict__ O) {
  constexpr int S = 2048, D = 512, DK = 64;
  constexpr int LDK = 72;             // padded row stride (144B)
  __shared__ bf16 Ks[2][32 * LDK];    // [key][d] row-major, double buffered
  __shared__ bf16 Vs[2][32 * LDK];    // [key][d] row-major, double buffered
  __shared__ bf16 Ps[4][16 * 48];     // per-wave probability tile [qrow][key]

  const int tid  = threadIdx.x;
  const int lane = tid & 31, wave = tid >> 5;
  const int lh = lane >> 4, ln = lane & 15;
  const int b = blockIdx.z, h = blockIdx.y;
  const int qbase = blockIdx.x * 64 + wave * 16;

  const int lkey = tid >> 2, ldc = (tid & 3) * 16;   // K/V loader mapping
  auto issue_kv = [&](int kb, int bufi) {
    const bf16* kp = K + ((size_t)(b * S + kb + lkey)) * D + h * DK + ldc;
    const bf16* vp = V + ((size_t)(b * S + kb + lkey)) * D + h * DK + ldc;
    stage_b128(kp,     &Ks[bufi][lkey * LDK + ldc]);
    stage_b128(kp + 8, &Ks[bufi][lkey * LDK + ldc + 8]);
    stage_b128(vp,     &Vs[bufi][lkey * LDK + ldc]);
    stage_b128(vp + 8, &Vs[bufi][lkey * LDK + ldc + 8]);
  };

  // Q fragments: 16 rows x 64 d as two 16x32 A-fragments
  FragAB qa[2];
  {
    const bf16* qp = Q + ((size_t)(b * S + qbase + ln)) * D + h * DK;
#pragma unroll
    for (int dc = 0; dc < 2; dc++) {
      qa[dc].q[0] = *(const u32x4*)(qp + dc * 32 + 16 * lh);
      qa[dc].q[1] = *(const u32x4*)(qp + dc * 32 + 16 * lh + 8);
    }
  }

  float rmax[8], rsum[8];
  v8f zero = {};
  v8f o[4];
#pragma unroll
  for (int r = 0; r < 8; r++) { rmax[r] = -__builtin_inff(); rsum[r] = 0.f; }
#pragma unroll
  for (int nt = 0; nt < 4; nt++) o[nt] = zero;

  issue_kv(0, 0);
  stage_wait();
  __syncthreads();

  int buf = 0;
  for (int kb = 0; kb < S; kb += 32) {
    if (kb + 32 < S) issue_kv(kb + 32, buf ^ 1);  // overlap with compute

    // scores: two 16(q) x 16(key) tiles, reduction over d = 64 (2 WMMAs each)
    v8f sc[2];
#pragma unroll
    for (int ks = 0; ks < 2; ks++) {
      v8f c = zero;
#pragma unroll
      for (int dc = 0; dc < 2; dc++) {
        FragAB kf;
        const bf16* kk = &Ks[buf][(ks * 16 + ln) * LDK + dc * 32 + 16 * lh];
        kf.q[0] = *(const u32x4*)kk;
        kf.q[1] = *(const u32x4*)(kk + 8);
        c = WMMA_BF16(qa[dc].v, kf.v, c);
      }
      sc[ks] = c;
    }

    // floor(s/8) + online softmax update; C layout: row = r + 8*lh, key = ln
    bf16* pp = Ps[wave];
#pragma unroll
    for (int r = 0; r < 8; r++) {
      float s0 = floorf(sc[0][r] * 0.125f);
      float s1 = floorf(sc[1][r] * 0.125f);
      float tm = fmaxf(s0, s1);
#pragma unroll
      for (int m = 1; m < 16; m <<= 1) tm = fmaxf(tm, __shfl_xor(tm, m, 32));
      float nm    = fmaxf(rmax[r], tm);
      float alpha = __expf(rmax[r] - nm);
      float p0 = __expf(s0 - nm);
      float p1 = __expf(s1 - nm);
      float ts = p0 + p1;
#pragma unroll
      for (int m = 1; m < 16; m <<= 1) ts += __shfl_xor(ts, m, 32);
      rsum[r] = rsum[r] * alpha + ts;
      rmax[r] = nm;
#pragma unroll
      for (int nt = 0; nt < 4; nt++) o[nt][r] *= alpha;
      int prow = r + 8 * lh;
      pp[prow * 48 + ln]      = (bf16)p0;
      pp[prow * 48 + 16 + ln] = (bf16)p1;
    }
    asm volatile("s_wait_dscnt 0" ::: "memory");

    // O += P(16x32) @ V(32x64): one A-fragment, 4 WMMAs over DK columns,
    // V fragments gathered column-wise from the row-major tile
    FragAB pa;
    pa.q[0] = *(const u32x4*)&pp[ln * 48 + 16 * lh];
    pa.q[1] = *(const u32x4*)&pp[ln * 48 + 16 * lh + 8];
#pragma unroll
    for (int nt = 0; nt < 4; nt++) {
      FragAB vf;
      const bf16* vv = &Vs[buf][(16 * lh) * LDK + nt * 16 + ln];  // key = 16*lh + e
#pragma unroll
      for (int e = 0; e < 16; e++) vf.h[e] = vv[e * LDK];
      o[nt] = WMMA_BF16(pa.v, vf.v, o[nt]);
    }
    stage_wait();
    __syncthreads();
    buf ^= 1;
  }

  // normalize and store bf16 (feeds Wo projection GEMM)
#pragma unroll
  for (int r = 0; r < 8; r++) {
    float inv = 1.0f / rsum[r];
    int row = qbase + r + 8 * lh;
#pragma unroll
    for (int nt = 0; nt < 4; nt++)
      O[((size_t)(b * S + row)) * D + h * DK + nt * 16 + ln] = (bf16)(o[nt][r] * inv);
  }
}

// ---------------------------------------------------------------------------
// Fused residual-add + LayerNorm (biased variance, eps=1e-5), D=512.
// One block per row; also emits bf16 copy for the next GEMM if outb != null.
// ---------------------------------------------------------------------------
__global__ __launch_bounds__(256) void add_ln_kernel(
    const float* __restrict__ resid, const float* __restrict__ src,
    const float* __restrict__ g, const float* __restrict__ bt,
    float* __restrict__ outf, bf16* __restrict__ outb) {
  constexpr int D = 512;
  __shared__ float s_sum[8], s_sq[8];
  __shared__ float s_mean, s_rstd;
  const int row = blockIdx.x, t = threadIdx.x;
  const size_t base = (size_t)row * D;
  float v0 = resid[base + t]       + src[base + t];
  float v1 = resid[base + t + 256] + src[base + t + 256];
  float sum = v0 + v1, sq = v0 * v0 + v1 * v1;
#pragma unroll
  for (int m = 1; m < 32; m <<= 1) {
    sum += __shfl_xor(sum, m, 32);
    sq  += __shfl_xor(sq, m, 32);
  }
  int wv = t >> 5, lane = t & 31;
  if (lane == 0) { s_sum[wv] = sum; s_sq[wv] = sq; }
  __syncthreads();
  if (t == 0) {
    float S1 = 0.f, S2 = 0.f;
    for (int i = 0; i < 8; i++) { S1 += s_sum[i]; S2 += s_sq[i]; }
    float mean = S1 / (float)D;
    float var  = S2 / (float)D - mean * mean;
    s_mean = mean;
    s_rstd = rsqrtf(var + 1e-5f);
  }
  __syncthreads();
  float mean = s_mean, rstd = s_rstd;
  float r0 = (v0 - mean) * rstd * g[t]       + bt[t];
  float r1 = (v1 - mean) * rstd * g[t + 256] + bt[t + 256];
  outf[base + t]       = r0;
  outf[base + t + 256] = r1;
  if (outb) {
    outb[base + t]       = (bf16)r0;
    outb[base + t + 256] = (bf16)r1;
  }
}

// ---------------------------------------------------------------------------
// Host orchestration
// ---------------------------------------------------------------------------
extern "C" void kernel_launch(void* const* d_in, const int* in_sizes, int n_in,
                              void* d_out, int out_size, void* d_ws, size_t ws_size,
                              hipStream_t stream) {
  (void)in_sizes; (void)n_in; (void)out_size; (void)ws_size;
  constexpr int Bb = 2, S = 2048, D = 512, FF = 2048, M = Bb * S, H = 8;

  const float* x   = (const float*)d_in[0];
  const float* enc = (const float*)d_in[1];

  char* w = (char*)d_ws;
  size_t off = 0;
  auto alloc = [&](size_t bytes) -> void* {
    void* p = w + off;
    off += (bytes + 255) & ~(size_t)255;
    return p;
  };
  bf16* xb   = (bf16*)alloc((size_t)M * D * 2);    // bf16 of current activation
  bf16* encb = (bf16*)alloc((size_t)M * D * 2);
  bf16* wb[8];
  for (int i = 0; i < 8; i++) wb[i] = (bf16*)alloc((size_t)D * D * 2);
  bf16* w1b = (bf16*)alloc((size_t)D * FF * 2);
  bf16* w2b = (bf16*)alloc((size_t)FF * D * 2);
  bf16* Qb  = (bf16*)alloc((size_t)M * D * 2);
  bf16* Kb  = (bf16*)alloc((size_t)M * D * 2);
  bf16* Vb  = (bf16*)alloc((size_t)M * D * 2);
  bf16* Ob  = (bf16*)alloc((size_t)M * D * 2);
  float* Cf  = (float*)alloc((size_t)M * D * 4);   // GEMM f32 output
  float* x1f = (float*)alloc((size_t)M * D * 4);   // f32 residual stream
  bf16* F1b = (bf16*)alloc((size_t)M * FF * 2);    // FFN hidden (bf16)

  auto cvt = [&](const void* src, bf16* dst, int n) {
    cvt_f32_bf16<<<(n + 255) / 256, 256, 0, stream>>>((const float*)src, dst, n);
  };
  cvt(x, xb, M * D);
  cvt(enc, encb, M * D);
  const int widx[8] = {4, 6, 8, 10, 12, 14, 16, 18};  // sa Wq,Wk,Wv,Wo, ca Wq,Wk,Wv,Wo
  for (int i = 0; i < 8; i++) cvt(d_in[widx[i]], wb[i], D * D);
  cvt(d_in[20], w1b, D * FF);
  cvt(d_in[22], w2b, FF * D);

  auto gemm = [&](const bf16* A, const bf16* Wm, const float* bias, float* cf, bf16* cb,
                  int Mv, int Nv, int Kv, bool relu) {
    dim3 grid(Nv / 128, Mv / 128);
    if (relu)
      gemm_bias_kernel<true, true><<<grid, 256, 0, stream>>>(A, Wm, bias, cf, cb, Mv, Nv, Kv);
    else if (cb)
      gemm_bias_kernel<false, true><<<grid, 256, 0, stream>>>(A, Wm, bias, cf, cb, Mv, Nv, Kv);
    else
      gemm_bias_kernel<false, false><<<grid, 256, 0, stream>>>(A, Wm, bias, cf, cb, Mv, Nv, Kv);
  };
  dim3 agrid(S / 64, H, Bb);

  // --- self attention ---
  gemm(xb, wb[0], (const float*)d_in[5], nullptr, Qb, M, D, D, false);
  gemm(xb, wb[1], (const float*)d_in[7], nullptr, Kb, M, D, D, false);
  gemm(xb, wb[2], (const float*)d_in[9], nullptr, Vb, M, D, D, false);
  attn_kernel<<<agrid, 128, 0, stream>>>(Qb, Kb, Vb, Ob);
  gemm(Ob, wb[3], (const float*)d_in[11], Cf, nullptr, M, D, D, false);
  add_ln_kernel<<<M, 256, 0, stream>>>(x, Cf, (const float*)d_in[24], (const float*)d_in[25],
                                       x1f, xb);
  // --- cross attention ---
  gemm(xb,   wb[4], (const float*)d_in[13], nullptr, Qb, M, D, D, false);
  gemm(encb, wb[5], (const float*)d_in[15], nullptr, Kb, M, D, D, false);
  gemm(encb, wb[6], (const float*)d_in[17], nullptr, Vb, M, D, D, false);
  attn_kernel<<<agrid, 128, 0, stream>>>(Qb, Kb, Vb, Ob);
  gemm(Ob, wb[7], (const float*)d_in[19], Cf, nullptr, M, D, D, false);
  add_ln_kernel<<<M, 256, 0, stream>>>(x1f, Cf, (const float*)d_in[26], (const float*)d_in[27],
                                       x1f, xb);
  // --- feed-forward ---
  gemm(xb,  w1b, (const float*)d_in[21], nullptr, F1b, M, FF, D, true);
  gemm(F1b, w2b, (const float*)d_in[23], Cf, nullptr, M, D, FF, false);
  add_ln_kernel<<<M, 256, 0, stream>>>(x1f, Cf, (const float*)d_in[28], (const float*)d_in[29],
                                       (float*)d_out, nullptr);
}
